// _PointnetSAModuleBase_37503654428685
// MI455X (gfx1250) — matrix operand order repaired and checked by reading, not compiled, not run on hardware
//
#include <hip/hip_runtime.h>
#include <hip/hip_bf16.h>

#define BB 4
#define NN 16384
#define NPOINT 1024
#define NSAMPLE 32
#define CC 64
#define CIN_P 96            // 3+64=67 padded to 3*32
#define RADIUS2 0.01f
#define BIGF 1.0e10f

typedef __attribute__((ext_vector_type(16))) _Float16 v16h;
typedef __attribute__((ext_vector_type(8)))  _Float16 v8h;
typedef __attribute__((ext_vector_type(8)))  float    v8f;

// ---------------------------------------------------------------------------
// Kernel 0: convert weights to f16, padded layouts for WMMA B-fragments.
// wh layout (in halfs): [0,6144) W1h 64x96 | [6144,10240) W2h 64x64 |
//                       [10240,18432) W3h 128x64
// ---------------------------------------------------------------------------
__global__ void wcvt_kernel(const float* __restrict__ w1,
                            const float* __restrict__ w2,
                            const float* __restrict__ w3,
                            _Float16* __restrict__ wh) {
  int i = blockIdx.x * blockDim.x + threadIdx.x;
  if (i < 6144) {
    int o = i / CIN_P, c = i % CIN_P;
    wh[i] = (c < 67) ? (_Float16)w1[o * 67 + c] : (_Float16)0.0f;
  } else if (i < 10240) {
    wh[i] = (_Float16)w2[i - 6144];
  } else if (i < 18432) {
    wh[i] = (_Float16)w3[i - 10240];
  }
}

// ---------------------------------------------------------------------------
// Kernel 1: furthest point sampling. One block per batch, 1024 threads,
// xyz + running min-dist held in registers (16 points/thread).
// Writes new_xyz directly into d_out[0 .. B*NPOINT*3).
// ---------------------------------------------------------------------------
__global__ void fps_kernel(const float* __restrict__ xyz,
                           float* __restrict__ new_xyz) {
  const int b   = blockIdx.x;
  const int tid = threadIdx.x;               // 0..1023
  const float* X = xyz + (size_t)b * NN * 3;

  float px[16], py[16], pz[16], dist[16];
#pragma unroll
  for (int k = 0; k < 16; ++k) {
    int n = tid + (k << 10);
    px[k] = X[n * 3 + 0];
    py[k] = X[n * 3 + 1];
    pz[k] = X[n * 3 + 2];
    dist[k] = BIGF;
  }

  __shared__ float sval[32];
  __shared__ int   sidx[32];
  __shared__ int   sfar;

  int farthest = 0;
  for (int s = 0; s < NPOINT; ++s) {
    float cx = X[farthest * 3 + 0];
    float cy = X[farthest * 3 + 1];
    float cz = X[farthest * 3 + 2];
    if (tid == 0) {
      float* o = new_xyz + ((size_t)b * NPOINT + s) * 3;
      o[0] = cx; o[1] = cy; o[2] = cz;
    }
    float bestv = -1.0f;
    int   besti = 0;
#pragma unroll
    for (int k = 0; k < 16; ++k) {
      float dx = px[k] - cx, dy = py[k] - cy, dz = pz[k] - cz;
      float d  = dx * dx + dy * dy + dz * dz;
      float nd = fminf(dist[k], d);
      dist[k]  = nd;
      if (nd > bestv) { bestv = nd; besti = tid + (k << 10); }
    }
    // wave32 reduction (argmax, tie -> smaller index)
#pragma unroll
    for (int off = 16; off > 0; off >>= 1) {
      float ov = __shfl_down(bestv, off, 32);
      int   oi = __shfl_down(besti, off, 32);
      if (ov > bestv || (ov == bestv && oi < besti)) { bestv = ov; besti = oi; }
    }
    if ((tid & 31) == 0) { sval[tid >> 5] = bestv; sidx[tid >> 5] = besti; }
    __syncthreads();
    if (tid < 32) {
      bestv = sval[tid]; besti = sidx[tid];
#pragma unroll
      for (int off = 16; off > 0; off >>= 1) {
        float ov = __shfl_down(bestv, off, 32);
        int   oi = __shfl_down(besti, off, 32);
        if (ov > bestv || (ov == bestv && oi < besti)) { bestv = ov; besti = oi; }
      }
      if (tid == 0) sfar = besti;
    }
    __syncthreads();
    farthest = sfar;
  }
}

// ---------------------------------------------------------------------------
// Kernel 2: ball query. One wave per query point; ballot + popc ordered
// append gives exactly the first NSAMPLE indices (ascending) within radius.
// ---------------------------------------------------------------------------
__global__ void bq_kernel(const float* __restrict__ xyz,
                          const float* __restrict__ new_xyz,
                          int* __restrict__ nidx) {
  const int lane = threadIdx.x & 31;
  const int wv   = threadIdx.x >> 5;
  const int pg   = blockIdx.x * (blockDim.x >> 5) + wv;   // 0..B*NPOINT-1
  const int b    = pg >> 10;
  const float* X = xyz + (size_t)b * NN * 3;

  const float qx = new_xyz[pg * 3 + 0];
  const float qy = new_xyz[pg * 3 + 1];
  const float qz = new_xyz[pg * 3 + 2];

  __shared__ int slots[8][NSAMPLE];
  int* slot = slots[wv];
  slot[lane] = 0;

  int cnt = 0;
  for (int n0 = 0; n0 < NN && cnt < NSAMPLE; n0 += 32) {
    int n = n0 + lane;
    float dx = X[n * 3 + 0] - qx;
    float dy = X[n * 3 + 1] - qy;
    float dz = X[n * 3 + 2] - qz;
    bool in = (dx * dx + dy * dy + dz * dz) < RADIUS2;
    unsigned mask = (unsigned)__ballot(in);
    int rank = __popc(mask & ((1u << lane) - 1u));
    if (in) {
      int t = cnt + rank;
      if (t < NSAMPLE) slot[t] = n;
    }
    cnt += __popc(mask);
  }
  __syncthreads();
  int v = slot[lane];
  int out = (lane < cnt) ? v : slot[0];
  nidx[(size_t)pg * NSAMPLE + lane] = out;
}

// ---------------------------------------------------------------------------
// WMMA fragment helpers (wave32, 16-bit operands, 16x16x32)
// ---------------------------------------------------------------------------
static __device__ __forceinline__
v16h ld_afrag(const _Float16* H, int stride, int mt, int kc, int lane) {
  int row = mt * 16 + (lane & 15);
  int kb  = kc * 32 + ((lane & 16) ? 8 : 0);
  const _Float16* p = H + row * stride + kb;
  v8h lo = *(const v8h*)(p);
  v8h hi = *(const v8h*)(p + 16);
  return __builtin_shufflevector(lo, hi, 0,1,2,3,4,5,6,7,8,9,10,11,12,13,14,15);
}

static __device__ __forceinline__
v16h ld_bfrag(const _Float16* W, int stride, int nt, int kc, int lane) {
  int col = nt * 16 + (lane & 15);
  int kb  = kc * 32 + ((lane & 16) ? 8 : 0);
  const _Float16* p = W + col * stride + kb;
  v8h lo = *(const v8h*)(p);
  v8h hi = *(const v8h*)(p + 16);
  return __builtin_shufflevector(lo, hi, 0,1,2,3,4,5,6,7,8,9,10,11,12,13,14,15);
}

static __device__ __forceinline__
void st_tile_relu(_Float16* O, int stride, int mt, int nt, int lane,
                  v8f acc, float bias) {
  int n  = nt * 16 + (lane & 15);
  int mb = mt * 16 + ((lane & 16) ? 8 : 0);
#pragma unroll
  for (int r = 0; r < 8; ++r) {
    float v = acc[r] + bias;
    v = v > 0.0f ? v : 0.0f;
    O[(mb + r) * stride + n] = (_Float16)v;
  }
}

// ---------------------------------------------------------------------------
// Kernel 3: gather + 3-layer WMMA MLP + max-pool. One wave per query point,
// 2 waves per block. Dynamic LDS: f16 weights (shared) + per-wave tiles.
// ---------------------------------------------------------------------------
__global__ void mlp_kernel(const float* __restrict__ xyz,
                           const float* __restrict__ features,
                           const int* __restrict__ nidx,
                           const float* __restrict__ new_xyz,
                           const _Float16* __restrict__ wh,
                           const float* __restrict__ b1,
                           const float* __restrict__ b2,
                           const float* __restrict__ b3,
                           float* __restrict__ out_feat) {
  extern __shared__ _Float16 smem[];
  const int lane = threadIdx.x & 31;
  const int wv   = threadIdx.x >> 5;
  const int pg   = blockIdx.x * 2 + wv;      // 0..B*NPOINT-1
  const int b    = pg >> 10;
  const int p    = pg & 1023;

  // cooperative weight copy (f16, 9216 dwords)
  {
    const int* src = (const int*)wh;
    int* dst = (int*)smem;
    for (int i = threadIdx.x; i < 9216; i += blockDim.x) dst[i] = src[i];
  }
  __syncthreads();

  const _Float16* W1h = smem;             // 64 x 96
  const _Float16* W2h = smem + 6144;      // 64 x 64
  const _Float16* W3h = smem + 10240;     // 128 x 64
  _Float16* H1 = smem + 18432 + wv * 5120;  // 32 x 96 (also reused 32 x 64)
  _Float16* H2 = H1 + 3072;                 // 32 x 64

  // -------- gather: lane = sample --------
  {
    int idx = nidx[(size_t)pg * NSAMPLE + lane];
    const float* F = features + ((size_t)b * NN + idx) * CC;
    const float* P = xyz + ((size_t)b * NN + idx) * 3;
    __builtin_prefetch(F, 0, 1);
    float cx = new_xyz[pg * 3 + 0];
    float cy = new_xyz[pg * 3 + 1];
    float cz = new_xyz[pg * 3 + 2];
    _Float16* row = H1 + lane * CIN_P;
    row[0] = (_Float16)(P[0] - cx);
    row[1] = (_Float16)(P[1] - cy);
    row[2] = (_Float16)(P[2] - cz);
#pragma unroll
    for (int c = 0; c < CC; c += 4) {
      float4 f = *(const float4*)(F + c);
      row[3 + c + 0] = (_Float16)f.x;
      row[3 + c + 1] = (_Float16)f.y;
      row[3 + c + 2] = (_Float16)f.z;
      row[3 + c + 3] = (_Float16)f.w;
    }
#pragma unroll
    for (int c = 67; c < CIN_P; ++c) row[c] = (_Float16)0.0f;
  }
  // per-wave tile: LDS ops from one wave are in-order; no barrier needed.

  // -------- layer 1: (32x96) x (96x64) -> H2 --------
#pragma unroll
  for (int mt = 0; mt < 2; ++mt) {
#pragma unroll
    for (int nt = 0; nt < 4; ++nt) {
      v8f acc = {};
#pragma unroll
      for (int kc = 0; kc < 3; ++kc) {
        v16h a = ld_afrag(H1, CIN_P, mt, kc, lane);
        v16h w = ld_bfrag(W1h, CIN_P, nt, kc, lane);
        acc = __builtin_amdgcn_wmma_f32_16x16x32_f16(false, a, false, w,
                                                     (short)0, acc, false, false);
      }
      st_tile_relu(H2, 64, mt, nt, lane, acc, b1[nt * 16 + (lane & 15)]);
    }
  }

  // -------- layer 2: (32x64) x (64x64) -> H1 (reused, stride 64) --------
#pragma unroll
  for (int mt = 0; mt < 2; ++mt) {
#pragma unroll
    for (int nt = 0; nt < 4; ++nt) {
      v8f acc = {};
#pragma unroll
      for (int kc = 0; kc < 2; ++kc) {
        v16h a = ld_afrag(H2, 64, mt, kc, lane);
        v16h w = ld_bfrag(W2h, 64, nt, kc, lane);
        acc = __builtin_amdgcn_wmma_f32_16x16x32_f16(false, a, false, w,
                                                     (short)0, acc, false, false);
      }
      st_tile_relu(H1, 64, mt, nt, lane, acc, b2[nt * 16 + (lane & 15)]);
    }
  }

  // -------- layer 3: (32x64) x (64x128), bias+relu+max over samples --------
#pragma unroll
  for (int nt = 0; nt < 8; ++nt) {
    const int n = nt * 16 + (lane & 15);
    const float bias = b3[n];
    float vmax = 0.0f;                       // relu floor
#pragma unroll
    for (int mt = 0; mt < 2; ++mt) {
      v8f acc = {};
#pragma unroll
      for (int kc = 0; kc < 2; ++kc) {
        v16h a = ld_afrag(H1, 64, mt, kc, lane);
        v16h w = ld_bfrag(W3h, 64, nt, kc, lane);
        acc = __builtin_amdgcn_wmma_f32_16x16x32_f16(false, a, false, w,
                                                     (short)0, acc, false, false);
      }
#pragma unroll
      for (int r = 0; r < 8; ++r) vmax = fmaxf(vmax, acc[r] + bias);
    }
    vmax = fmaxf(vmax, __shfl_xor(vmax, 16, 32));   // fuse the two row halves
    if (lane < 16)
      out_feat[((size_t)b * 128 + n) * NPOINT + p] = vmax;
  }
}

// ---------------------------------------------------------------------------
extern "C" void kernel_launch(void* const* d_in, const int* in_sizes, int n_in,
                              void* d_out, int out_size, void* d_ws, size_t ws_size,
                              hipStream_t stream) {
  const float* xyz      = (const float*)d_in[0];
  const float* features = (const float*)d_in[1];
  const float* w1 = (const float*)d_in[2];
  const float* b1 = (const float*)d_in[3];
  const float* w2 = (const float*)d_in[4];
  const float* b2 = (const float*)d_in[5];
  const float* w3 = (const float*)d_in[6];
  const float* b3 = (const float*)d_in[7];

  float* out      = (float*)d_out;                 // [0,12288) new_xyz
  float* out_feat = out + BB * NPOINT * 3;         // (B,128,NPOINT)

  int* nidx = (int*)d_ws;                                        // 131072 ints
  _Float16* wh = (_Float16*)((char*)d_ws + (size_t)BB * NPOINT * NSAMPLE * sizeof(int));

  wcvt_kernel<<<72, 256, 0, stream>>>(w1, w2, w3, wh);
  fps_kernel<<<BB, 1024, 0, stream>>>(xyz, out);
  bq_kernel<<<(BB * NPOINT) / 8, 256, 0, stream>>>(xyz, out, nidx);

  const size_t lds_bytes = (18432 + 2 * 5120) * sizeof(_Float16);  // 57344
  mlp_kernel<<<(BB * NPOINT) / 2, 64, lds_bytes, stream>>>(
      xyz, features, nidx, out, wh, b1, b2, b3, out_feat);
}
